// RNN_15968688407300
// MI455X (gfx1250) — matrix operand order repaired and checked by reading, not compiled, not run on hardware
//
#include <hip/hip_runtime.h>
#include <hip/hip_bf16.h>
#include <math.h>

#define B_   32
#define S_   512
#define V_   32000
#define H_   1024
#define C_   2
#define NWG  64          // workgroups; each owns 16 output columns
#define COLS 16
#define TPB  256         // 8 wave32 waves

typedef __attribute__((ext_vector_type(16))) __bf16 bf16x16;
typedef __attribute__((ext_vector_type(8)))  float  f32x8;

union FragAB {
    bf16x16        v;
    uint4          q[2];
    unsigned short us[16];
};

__device__ __forceinline__ unsigned short f2bf(float f) {
    union { float f; unsigned u; } c; c.f = f;
    unsigned u = c.u;
    unsigned r = u + 0x7FFFu + ((u >> 16) & 1u);   // round-to-nearest-even
    return (unsigned short)(r >> 16);
}
__device__ __forceinline__ float bf2f(unsigned short h) {
    union { unsigned u; float f; } c; c.u = ((unsigned)h) << 16;
    return c.f;
}

__global__ void rnn_init(unsigned short* h0, unsigned* bar) {
    int tid = blockIdx.x * blockDim.x + threadIdx.x;
    for (int i = tid; i < B_ * H_; i += gridDim.x * blockDim.x) h0[i] = 0;
    if (tid == 0) { bar[0] = 0u; bar[1] = 0u; }
}

// Persistent kernel: one WG per WGP, 2 waves/SIMD. Cap the occupancy target there
// so the VGPR budget is large enough to keep all 8 A-fragments live at once.
__global__ __attribute__((amdgpu_waves_per_eu(2))) __launch_bounds__(TPB)
void rnn_persistent(const int* __restrict__ x,
                    const float* __restrict__ Wxh,
                    const float* __restrict__ bxh,
                    const float* __restrict__ Whh,
                    const float* __restrict__ bhh,
                    const float* __restrict__ fcw,
                    const float* __restrict__ fcb,
                    float* __restrict__ out,
                    unsigned short* __restrict__ h0buf,
                    unsigned short* __restrict__ h1buf,
                    unsigned* __restrict__ bar)
{
    __shared__ float etile[2][B_][COLS];          // async-gathered embeddings (dbl buf)
    __shared__ float red[6][TPB];                 // K-split partial sums
    __shared__ float bsum[COLS];                  // b_xh + b_hh folded

    const int wg      = blockIdx.x;
    const int tid     = threadIdx.x;
    const int lane    = tid & 31;
    const int wv      = tid >> 5;
    const int colbase = wg * COLS;

    const int tileM  = (wv >> 2) << 4;  // waves 0-3 -> M 0..15, waves 4-7 -> M 16..31
    const int wk     = wv & 3;          // K quarter owned by this wave
    const int n      = lane & 15;
    const int laneHi = lane >> 4;
    const int kbeg   = wk * 256;

    if (tid < COLS) bsum[tid] = bxh[colbase + tid] + bhh[colbase + tid];

    // ---- one-time: B (weight) fragments -> registers, resident for all 512 steps ----
    // B frag layout: lane holds column n = lane%16; K = k0 + laneHi*16 .. +15 contiguous
    FragAB bw[8];
    {
        const float* wrow = Whh + (size_t)(colbase + n) * H_;
        #pragma unroll
        for (int i = 0; i < 8; ++i) {
            int kb = kbeg + i * 32 + (laneHi << 4);
            #pragma unroll
            for (int e = 0; e < 16; e += 4) {
                float4 w4 = *(const float4*)(wrow + kb + e);
                bw[i].us[e + 0] = f2bf(w4.x);
                bw[i].us[e + 1] = f2bf(w4.y);
                bw[i].us[e + 2] = f2bf(w4.z);
                bw[i].us[e + 3] = f2bf(w4.w);
            }
        }
    }
    __syncthreads();

    unsigned* cnt  = bar;
    unsigned* flag = bar + 1;

    // ---- issue async gather of e_0 before the loop ----
    #pragma unroll
    for (int rep = 0; rep < 2; ++rep) {
        int idx = tid + rep * TPB;
        int b = idx >> 4, nn = idx & 15;
        int tok = x[b * S_ + 0];
        const float* gp = Wxh + (size_t)(colbase + nn) * V_ + (size_t)tok;
        unsigned ldsoff = (unsigned)(unsigned long long)&etile[0][b][nn];
        asm volatile("global_load_async_to_lds_b32 %0, %1, off"
                     :: "v"(ldsoff), "v"((unsigned long long)gp) : "memory");
    }

    for (int t = 0; t < S_; ++t) {
        const unsigned short* hprev = (t & 1) ? h1buf : h0buf;
        unsigned short*       hnext = (t & 1) ? h0buf : h1buf;
        const int par = t & 1;

        // ---- A fragments: ALL 16 b128 loads must issue before any WMMA ----
        const unsigned short* hrow = hprev + (size_t)(tileM + n) * H_;
        FragAB af[8];
        #pragma unroll
        for (int i = 0; i < 8; ++i) {
            int kb = kbeg + i * 32 + (laneHi << 4);   // elems 0-15 -> K kb..kb+15 contiguous
            af[i].q[0] = *(const uint4*)(hrow + kb);
            af[i].q[1] = *(const uint4*)(hrow + kb + 8);
        }
        // Hard scheduling fence: loads stay above, WMMAs below. Forces all 8
        // fragments live -> progressive partial s_wait_loadcnt instead of
        // serialized load->wait(0)->wmma pairs.
        __builtin_amdgcn_sched_barrier(0);

        f32x8 acc0 = {0.f,0.f,0.f,0.f,0.f,0.f,0.f,0.f};
        f32x8 acc1 = {0.f,0.f,0.f,0.f,0.f,0.f,0.f,0.f};
        #pragma unroll
        for (int i = 0; i < 8; i += 2) {   // two independent accumulation chains
            acc0 = __builtin_amdgcn_wmma_f32_16x16x32_bf16(false, af[i].v,     false, bw[i].v,
                                                           (short)0, acc0, false, false);
            acc1 = __builtin_amdgcn_wmma_f32_16x16x32_bf16(false, af[i + 1].v, false, bw[i + 1].v,
                                                           (short)0, acc1, false, false);
        }

        // wait for this wave's async e-gather (issued one step ahead)
        asm volatile("s_wait_asynccnt 0x0" ::: "memory");

        // ---- cross-wave K reduction through LDS ----
        if (wk != 0) {
            const int slot = (wv >= 4) ? (wv - 5 + 3) : (wv - 1);
            #pragma unroll
            for (int i = 0; i < 8; ++i) red[slot][lane * 8 + i] = acc0[i] + acc1[i];
        }
        __syncthreads();

        // ---- prefetch next step's embedding gather (hidden under barrier + WMMA) ----
        if (t + 1 < S_) {
            #pragma unroll
            for (int rep = 0; rep < 2; ++rep) {
                int idx = tid + rep * TPB;
                int b = idx >> 4, nn = idx & 15;
                int tok = x[b * S_ + (t + 1)];
                const float* gp = Wxh + (size_t)(colbase + nn) * V_ + (size_t)tok;
                unsigned ldsoff = (unsigned)(unsigned long long)&etile[par ^ 1][b][nn];
                asm volatile("global_load_async_to_lds_b32 %0, %1, off"
                             :: "v"(ldsoff), "v"((unsigned long long)gp) : "memory");
            }
        }

        if (wk == 0) {
            float accf[8];
            #pragma unroll
            for (int i = 0; i < 8; ++i) accf[i] = acc0[i] + acc1[i];
            const int sbase = (wv >= 4) ? 3 : 0;
            #pragma unroll
            for (int p = 0; p < 3; ++p)
                #pragma unroll
                for (int i = 0; i < 8; ++i) accf[i] += red[sbase + p][lane * 8 + i];
            // C/D layout: VGPR i -> M = tileM + i (+8 for lanes 16..31), N = lane%16
            #pragma unroll
            for (int i = 0; i < 8; ++i) {
                int m = tileM + i + (laneHi << 3);
                float v = tanhf(accf[i] + etile[par][m][n] + bsum[n]);
                hnext[(size_t)m * H_ + colbase + n] = f2bf(v);
            }
        }

        // ---- grid-wide release/acquire barrier (64 arrivals) ----
        __threadfence();
        __syncthreads();
        if (tid == 0) {
            unsigned arrived = __hip_atomic_fetch_add(cnt, 1u, __ATOMIC_ACQ_REL,
                                                      __HIP_MEMORY_SCOPE_AGENT);
            if (arrived == NWG - 1) {
                __hip_atomic_store(cnt, 0u, __ATOMIC_RELAXED, __HIP_MEMORY_SCOPE_AGENT);
                __hip_atomic_store(flag, (unsigned)(t + 1), __ATOMIC_RELEASE,
                                   __HIP_MEMORY_SCOPE_AGENT);
            } else {
                while (__hip_atomic_load(flag, __ATOMIC_ACQUIRE,
                                         __HIP_MEMORY_SCOPE_AGENT) < (unsigned)(t + 1))
                    __builtin_amdgcn_s_sleep(1);
            }
        }
        __syncthreads();
    }

    // ---- final FC: out[b,c] = h_last[b,:] . fc_w[c,:] + fc_b[c]  (t=511 wrote buf0) ----
    if (wg == 0 && tid < B_ * C_) {
        const unsigned short* hf = h0buf;
        int b = tid >> 1, c = tid & 1;
        float a = fcb[c];
        for (int j = 0; j < H_; ++j)
            a += bf2f(hf[b * H_ + j]) * fcw[c * H_ + j];
        out[b * C_ + c] = a;
    }
}

extern "C" void kernel_launch(void* const* d_in, const int* in_sizes, int n_in,
                              void* d_out, int out_size, void* d_ws, size_t ws_size,
                              hipStream_t stream) {
    const int*   x   = (const int*)  d_in[0];
    const float* Wxh = (const float*)d_in[1];
    const float* bxh = (const float*)d_in[2];
    const float* Whh = (const float*)d_in[3];
    const float* bhh = (const float*)d_in[4];
    const float* fcw = (const float*)d_in[5];
    const float* fcb = (const float*)d_in[6];
    float* out = (float*)d_out;

    // workspace layout: h ping-pong buffers (bf16) + barrier state  (~128 KB + 8 B)
    unsigned short* h0 = (unsigned short*)d_ws;
    unsigned short* h1 = h0 + B_ * H_;
    unsigned* bar      = (unsigned*)(h1 + B_ * H_);

    rnn_init<<<8, 256, 0, stream>>>(h0, bar);
    rnn_persistent<<<NWG, TPB, 0, stream>>>(x, Wxh, bxh, Whh, bhh, fcw, fcb,
                                            out, h0, h1, bar);
}
